// Transformer_62964220559987
// MI455X (gfx1250) — matrix operand order repaired and checked by reading, not compiled, not run on hardware
//
#include <hip/hip_runtime.h>
#include <hip/hip_bf16.h>

// ---- CDNA5 WMMA types (wave32) ----
typedef __attribute__((ext_vector_type(16))) _Float16 v16h;
typedef __attribute__((ext_vector_type(8)))  _Float16 v8h;
typedef __attribute__((ext_vector_type(8)))  float    v8f;

#define NTHREADS 256
#define NWAVES   8
#define EPSC     1e-5f
#define SCALEC   0.35355339059327373f   // (HEAD_DIM/HEADS)^-0.5 = 8^-0.5

// A operand (16-bit A 16x32 layout): two contiguous 16B ds loads per lane.
__device__ __forceinline__ v16h ldA(const _Float16* buf, int ld, int row0, int k0) {
  const int lane = threadIdx.x & 31;
  const int r    = row0 + (lane & 15);
  const int kh   = k0 + ((lane & 16) ? 16 : 0);
  const _Float16* p = buf + r * ld + kh;
  v8h lo = *(const v8h*)(p);
  v8h hi = *(const v8h*)(p + 8);
  return __builtin_shufflevector(lo, hi, 0,1,2,3,4,5,6,7,8,9,10,11,12,13,14,15);
}

// B operand (16-bit B 32x16 layout) from [n][k] k-fastest storage:
// two contiguous 16B ds loads per lane.
__device__ __forceinline__ v16h ldB(const _Float16* buf, int ld, int n0, int k0) {
  const int lane = threadIdx.x & 31;
  const int nA   = n0 + (lane & 15);
  const int kh   = k0 + ((lane & 16) ? 16 : 0);
  const _Float16* p = buf + nA * ld + kh;
  v8h lo = *(const v8h*)(p);
  v8h hi = *(const v8h*)(p + 8);
  return __builtin_shufflevector(lo, hi, 0,1,2,3,4,5,6,7,8,9,10,11,12,13,14,15);
}

// B operand straight from row-major f32 weights in global (L2-hot).
__device__ __forceinline__ v16h ldBglobal(const float* base, int ld, int n0, int k0) {
  const int lane = threadIdx.x & 31;
  const int nA   = n0 + (lane & 15);
  const int kh   = k0 + ((lane & 16) ? 16 : 0);
  v16h b;
#pragma unroll
  for (int e = 0; e < 16; ++e) b[e] = (_Float16)base[(kh + e) * ld + nA];
  return b;
}

__device__ __forceinline__ v8f wmma_f16(v16h a, v16h b, v8f c) {
  return __builtin_amdgcn_wmma_f32_16x16x32_f16(false, a, false, b,
                                                (short)0, c, false, false);
}

// ------------------------------------------------------------------
// Fused 4-layer mini-ViT: one workgroup (8 wave32) per batch sample.
// ------------------------------------------------------------------
__global__ __launch_bounds__(NTHREADS)
void vit_fused_kernel(const float* __restrict__ x,
                      const float* __restrict__ w_qkv, const float* __restrict__ b_qkv,
                      const float* __restrict__ w_o,   const float* __restrict__ b_o,
                      const float* __restrict__ ln1_g, const float* __restrict__ ln1_b,
                      const float* __restrict__ ln2_g, const float* __restrict__ ln2_b,
                      const float* __restrict__ w1,    const float* __restrict__ b1,
                      const float* __restrict__ w2,    const float* __restrict__ b2,
                      float* __restrict__ out) {
  // 126,336 B static LDS -> 2 workgroups per 320KB WGP
  __shared__ __align__(32) float    Xs[81 * 64];      // residual stream (f32)
  __shared__ __align__(32) _Float16 Y16[96 * 64];     // LN out / attn out (padded rows)
  __shared__ __align__(32) _Float16 QKV16[81 * 192];  // q|k|v
  __shared__ __align__(32) float    Sbuf[96 * 96];    // scores / crop scratch
  __shared__ __align__(32) _Float16 PQK[96 * 96];     // alias: P16 | (Qh,Kh) | (W1T,W2T)
  __shared__ __align__(32) _Float16 HV[96 * 32];      // alias: H16 | VhT
  __shared__ float rowM[96];
  __shared__ float rowS[96];

  _Float16* const P16 = PQK;               // 96x96
  _Float16* const Qh  = PQK;               // 96x32 (k fastest)
  _Float16* const Kh  = PQK + 96 * 32;     // 96x32 (k fastest)
  _Float16* const W1T = PQK;               // 16x64 [c][k], cols>=8 zero
  _Float16* const W2T = PQK + 16 * 64;     // 64x32 [c][k], k>=8 zero
  _Float16* const H16 = HV;                // 96x32, cols>=16 zero
  _Float16* const VhT = HV;                // 16x96 [c][k]

  const int b    = blockIdx.x;
  const int tid  = threadIdx.x;
  const int lane = tid & 31;
  const int wid  = tid >> 5;

  for (int idx = tid; idx < 81 * 64; idx += NTHREADS)
    Xs[idx] = x[(size_t)b * (81 * 64) + idx];
  __syncthreads();

  const int pn[4] = {9, 7, 5, 3};

#pragma unroll 1
  for (int layer = 0; layer < 4; ++layer) {
    const int p  = pn[layer];
    const int n  = p * p;
    const int Mt = (n + 15) >> 4;

    const float* Wq  = w_qkv + layer * 64 * 192;
    const float* Bq  = b_qkv + layer * 192;
    const float* Wo  = w_o   + layer * 64 * 64;
    const float* Bo  = b_o   + layer * 64;
    const float* G1  = ln1_g + layer * 64;
    const float* Be1 = ln1_b + layer * 64;
    const float* G2  = ln2_g + layer * 64;
    const float* Be2 = ln2_b + layer * 64;
    const float* Wm1 = w1 + layer * 64 * 8;
    const float* Bm1 = b1 + layer * 8;
    const float* Wm2 = w2 + layer * 8 * 64;
    const float* Bm2 = b2 + layer * 64;

    if (lane == 0) {                 // pull this layer's weights toward L2
      __builtin_prefetch(Wq + wid * 1536, 0, 3);
      __builtin_prefetch(Wo + wid * 512, 0, 3);
    }

    // ---------- LN1 -> Y16 (pad rows zeroed) ----------
    for (int r = tid; r < 96; r += NTHREADS) {
      if (r < n) {
        const float* xr = Xs + r * 64;
        float m = 0.f;
        for (int c = 0; c < 64; ++c) m += xr[c];
        m *= (1.f / 64.f);
        float v = 0.f;
        for (int c = 0; c < 64; ++c) { const float d = xr[c] - m; v += d * d; }
        v *= (1.f / 64.f);
        const float inv = rsqrtf(v + EPSC);
        for (int c = 0; c < 64; ++c)
          Y16[r * 64 + c] = (_Float16)((xr[c] - m) * inv * G1[c] + Be1[c]);
      } else {
        for (int c = 0; c < 64; ++c) Y16[r * 64 + c] = (_Float16)0.f;
      }
    }
    __syncthreads();

    // ---------- qkv = Y @ Wq + Bq   (n x 64) @ (64 x 192) ----------
    for (int t = wid; t < Mt * 12; t += NWAVES) {
      const int row0 = (t / 12) * 16, n0 = (t % 12) * 16;
      v8f acc = {};
#pragma unroll
      for (int kk = 0; kk < 2; ++kk)
        acc = wmma_f16(ldA(Y16, 64, row0, kk * 32), ldBglobal(Wq, 192, n0, kk * 32), acc);
      const int col = n0 + (lane & 15);
      const int mb  = (lane & 16) ? 8 : 0;
      const float bias = Bq[col];
      if (row0 + 16 <= n) {                       // full tile: branch-free stores
#pragma unroll
        for (int r = 0; r < 8; ++r)
          QKV16[(row0 + mb + r) * 192 + col] = (_Float16)(acc[r] + bias);
      } else {                                    // tail tile
#pragma unroll
        for (int r = 0; r < 8; ++r) {
          const int row = row0 + mb + r;
          if (row < n) QKV16[row * 192 + col] = (_Float16)(acc[r] + bias);
        }
      }
    }
    __syncthreads();

    // ---------- attention: 8 heads of head-dim 8 (zero-padded to K=32) ----------
#pragma unroll 1
    for (int h = 0; h < 8; ++h) {
      // stage padded per-head operands (branchless clamp+select)
      for (int idx = tid; idx < 96 * 32; idx += NTHREADS) {
        const int r = idx >> 5, k = idx & 31;
        const int rc = (r < n) ? r : (n - 1);
        const int kc = (k < 8) ? k : 7;
        const bool ok = (r < n) & (k < 8);
        const _Float16 qv = QKV16[rc * 192 + h * 8 + kc];
        const _Float16 kv = QKV16[rc * 192 + 64 + h * 8 + kc];
        Qh[idx] = ok ? qv : (_Float16)0.f;
        Kh[idx] = ok ? kv : (_Float16)0.f;
      }
      for (int idx = tid; idx < 16 * 96; idx += NTHREADS) {
        const int c = idx / 96, k = idx - c * 96;
        const int cc = (c < 8) ? c : 7;
        const int kc = (k < n) ? k : (n - 1);
        const _Float16 vv = QKV16[kc * 192 + 128 + h * 8 + cc];
        VhT[idx] = ((c < 8) & (k < n)) ? vv : (_Float16)0.f;
      }
      __syncthreads();

      // S = (q_h @ k_h^T) * SCALE   -- unconditional stores (Sbuf padded)
      for (int t = wid; t < Mt * Mt; t += NWAVES) {
        const int row0 = (t / Mt) * 16, n0 = (t % Mt) * 16;
        v8f acc = {};
        acc = wmma_f16(ldA(Qh, 32, row0, 0), ldB(Kh, 32, n0, 0), acc);
        const int col = n0 + (lane & 15);
        const int mb  = (lane & 16) ? 8 : 0;
#pragma unroll
        for (int r = 0; r < 8; ++r)
          Sbuf[(row0 + mb + r) * 96 + col] = acc[r] * SCALEC;
      }
      __syncthreads();

      // softmax phase 1: per-row max & 1/sum
      for (int r = tid; r < n; r += NTHREADS) {
        const float* sr = Sbuf + r * 96;
        float mx = -3.0e38f;
        for (int c = 0; c < n; ++c) mx = fmaxf(mx, sr[c]);
        float sum = 0.f;
        for (int c = 0; c < n; ++c) sum += __expf(sr[c] - mx);
        rowM[r] = mx;
        rowS[r] = 1.f / sum;
      }
      __syncthreads();

      // softmax phase 2: branchless elementwise probs over FULL 96x96
      for (int idx = tid; idx < 96 * 96; idx += NTHREADS) {
        const int r = idx / 96, c = idx - r * 96;
        const int rc = (r < n) ? r : (n - 1);
        const int cc = (c < n) ? c : (n - 1);
        const float e = __expf(Sbuf[rc * 96 + cc] - rowM[rc]) * rowS[rc];
        P16[idx] = ((r < n) & (c < n)) ? (_Float16)e : (_Float16)0.f;
      }
      __syncthreads();

      // out_h = P @ v_h
      const int Kt = (n + 31) >> 5;
      for (int t = wid; t < Mt; t += NWAVES) {
        const int row0 = t * 16;
        v8f acc = {};
        for (int kk = 0; kk < Kt; ++kk)
          acc = wmma_f16(ldA(P16, 96, row0, kk * 32), ldB(VhT, 96, 0, kk * 32), acc);
        const int cl = lane & 15;
        const int mb = (lane & 16) ? 8 : 0;
        if (cl < 8) {                       // one lane-mask set for whole block
          if (row0 + 16 <= n) {
#pragma unroll
            for (int r = 0; r < 8; ++r)
              Y16[(row0 + mb + r) * 64 + h * 8 + cl] = (_Float16)acc[r];
          } else {
#pragma unroll
            for (int r = 0; r < 8; ++r) {
              const int row = row0 + mb + r;
              if (row < n) Y16[row * 64 + h * 8 + cl] = (_Float16)acc[r];
            }
          }
        }
      }
      __syncthreads();
    }

    // ---------- proj: X += AO @ Wo + Bo ----------
    for (int t = wid; t < Mt * 4; t += NWAVES) {
      const int row0 = (t / 4) * 16, n0 = (t % 4) * 16;
      v8f acc = {};
#pragma unroll
      for (int kk = 0; kk < 2; ++kk)
        acc = wmma_f16(ldA(Y16, 64, row0, kk * 32), ldBglobal(Wo, 64, n0, kk * 32), acc);
      const int col = n0 + (lane & 15);
      const int mb  = (lane & 16) ? 8 : 0;
      const float bias = Bo[col];
      if (row0 + 16 <= n) {
#pragma unroll
        for (int r = 0; r < 8; ++r)
          Xs[(row0 + mb + r) * 64 + col] += acc[r] + bias;
      } else {
#pragma unroll
        for (int r = 0; r < 8; ++r) {
          const int row = row0 + mb + r;
          if (row < n) Xs[row * 64 + col] += acc[r] + bias;
        }
      }
    }
    __syncthreads();

    // ---------- LN2 -> Y16 ; stage padded W1T/W2T ; zero H16 pads ----------
    for (int r = tid; r < 96; r += NTHREADS) {
      if (r < n) {
        const float* xr = Xs + r * 64;
        float m = 0.f;
        for (int c = 0; c < 64; ++c) m += xr[c];
        m *= (1.f / 64.f);
        float v = 0.f;
        for (int c = 0; c < 64; ++c) { const float d = xr[c] - m; v += d * d; }
        v *= (1.f / 64.f);
        const float inv = rsqrtf(v + EPSC);
        for (int c = 0; c < 64; ++c)
          Y16[r * 64 + c] = (_Float16)((xr[c] - m) * inv * G2[c] + Be2[c]);
      } else {
        for (int c = 0; c < 64; ++c) Y16[r * 64 + c] = (_Float16)0.f;
      }
    }
    for (int idx = tid; idx < 16 * 64; idx += NTHREADS) {   // W1T[c][k]
      const int c = idx >> 6, k = idx & 63;
      W1T[idx] = (c < 8) ? (_Float16)Wm1[k * 8 + c] : (_Float16)0.f;
    }
    for (int idx = tid; idx < 64 * 32; idx += NTHREADS) {   // W2T[c][k]
      const int c = idx >> 5, k = idx & 31;
      W2T[idx] = (k < 8) ? (_Float16)Wm2[k * 64 + c] : (_Float16)0.f;
    }
    for (int idx = tid; idx < 96 * 32; idx += NTHREADS) H16[idx] = (_Float16)0.f;
    __syncthreads();

    // ---------- MLP1 + exact GELU: H = gelu(Y @ W1 + b1) ----------
    for (int t = wid; t < Mt; t += NWAVES) {
      const int row0 = t * 16;
      v8f acc = {};
#pragma unroll
      for (int kk = 0; kk < 2; ++kk)
        acc = wmma_f16(ldA(Y16, 64, row0, kk * 32), ldB(W1T, 64, 0, kk * 32), acc);
      const int cl = lane & 15;
      const int mb = (lane & 16) ? 8 : 0;
      const float bias = (cl < 8) ? Bm1[cl] : 0.f;
      // unconditional: garbage in pad rows only reaches pad rows of Xs
#pragma unroll
      for (int r = 0; r < 8; ++r) {
        const float u = acc[r] + bias;
        const float g = 0.5f * u * (1.f + erff(u * 0.7071067811865476f));
        H16[(row0 + mb + r) * 32 + cl] = (_Float16)g;   // cols 8..15 -> gelu(0)=0
      }
    }
    __syncthreads();

    // ---------- MLP2 + residual: X += H @ W2 + b2 ----------
    for (int t = wid; t < Mt * 4; t += NWAVES) {
      const int row0 = (t / 4) * 16, n0 = (t % 4) * 16;
      v8f acc = {};
      acc = wmma_f16(ldA(H16, 32, row0, 0), ldB(W2T, 32, n0, 0), acc);
      const int col = n0 + (lane & 15);
      const int mb  = (lane & 16) ? 8 : 0;
      const float bias = Bm2[col];
      if (row0 + 16 <= n) {
#pragma unroll
        for (int r = 0; r < 8; ++r)
          Xs[(row0 + mb + r) * 64 + col] += acc[r] + bias;
      } else {
#pragma unroll
        for (int r = 0; r < 8; ++r) {
          const int row = row0 + mb + r;
          if (row < n) Xs[row * 64 + col] += acc[r] + bias;
        }
      }
    }
    __syncthreads();

    // ---------- crop border (layers 0..2) or emit transposed output ----------
    if (layer != 3) {
      const int np  = p - 2;
      const int cnt = np * np * 64;
      for (int idx = tid; idx < cnt; idx += NTHREADS) {
        const int d  = idx & 63;
        const int r2 = idx >> 6;
        const int ii = r2 / np, jj = r2 % np;
        Sbuf[idx] = Xs[((ii + 1) * p + (jj + 1)) * 64 + d];
      }
      __syncthreads();
      for (int idx = tid; idx < cnt; idx += NTHREADS) Xs[idx] = Sbuf[idx];
      __syncthreads();
    } else {
      // out[b][d][ii][jj] = x[b][ii*3+jj][d]
      for (int idx = tid; idx < 64 * 9; idx += NTHREADS) {
        const int pos = idx % 9, d = idx / 9;
        out[(size_t)b * 576 + d * 9 + pos] = Xs[pos * 64 + d];
      }
    }
  }
}

extern "C" void kernel_launch(void* const* d_in, const int* in_sizes, int n_in,
                              void* d_out, int out_size, void* d_ws, size_t ws_size,
                              hipStream_t stream) {
  const float* x     = (const float*)d_in[0];
  const float* w_qkv = (const float*)d_in[1];
  const float* b_qkv = (const float*)d_in[2];
  const float* w_o   = (const float*)d_in[3];
  const float* b_o   = (const float*)d_in[4];
  const float* ln1_g = (const float*)d_in[5];
  const float* ln1_b = (const float*)d_in[6];
  const float* ln2_g = (const float*)d_in[7];
  const float* ln2_b = (const float*)d_in[8];
  const float* w1    = (const float*)d_in[9];
  const float* b1    = (const float*)d_in[10];
  const float* w2    = (const float*)d_in[11];
  const float* b2    = (const float*)d_in[12];

  const int B = in_sizes[0] / (81 * 64);

  vit_fused_kernel<<<B, NTHREADS, 0, stream>>>(
      x, w_qkv, b_qkv, w_o, b_o, ln1_g, ln1_b, ln2_g, ln2_b,
      w1, b1, w2, b2, (float*)d_out);

  (void)n_in; (void)out_size; (void)d_ws; (void)ws_size;
}
